// VAE_AllHist_GLM_17806934409621
// MI455X (gfx1250) — matrix operand order repaired and checked by reading, not compiled, not run on hardware
//
#include <hip/hip_runtime.h>
#include <hip/hip_bf16.h>
#include <math.h>

typedef __attribute__((ext_vector_type(16))) _Float16 v16h;
typedef __attribute__((ext_vector_type(8)))  float    v8f;

#define T_TOT  100000
#define E_N    1000
#define I_N    250
#define SUB    20
#define TN     200
#define NB     9
#define KE_PAD 1024
#define KI_PAD 256
#define TILE   256
#define HIST   200

// ---------------------------------------------------------------------------
// Kernel 1: syn_e = S_e @ C_syn_e^T, syn_i = S_i @ C_syn_i^T via WMMA f16->f32
// One wave computes one 16-row tile, N=20 handled as two 16-col WMMA tiles.
// 0/1 data is exact in f16; f32 accumulation => bit-exact GEMM.
// ---------------------------------------------------------------------------

__device__ inline v16h load_a_full(const float* __restrict__ rowp, int kbase) {
  // 16B-aligned fast path (valid for S_e rows: stride 1000 floats, kbase % 8 == 0)
  const float4* p0 = (const float4*)(rowp + kbase);
  const float4* p1 = (const float4*)(rowp + kbase + 8);
  float4 f0 = p0[0], f1 = p0[1], f2 = p1[0], f3 = p1[1];
  v16h a;
  a[0]=(_Float16)f0.x; a[1]=(_Float16)f0.y; a[2]=(_Float16)f0.z; a[3]=(_Float16)f0.w;
  a[4]=(_Float16)f1.x; a[5]=(_Float16)f1.y; a[6]=(_Float16)f1.z; a[7]=(_Float16)f1.w;
  a[8]=(_Float16)f2.x; a[9]=(_Float16)f2.y; a[10]=(_Float16)f2.z; a[11]=(_Float16)f2.w;
  a[12]=(_Float16)f3.x; a[13]=(_Float16)f3.y; a[14]=(_Float16)f3.z; a[15]=(_Float16)f3.w;
  return a;
}

__device__ inline v16h load_a_guard(const float* __restrict__ rowp, int kbase, int K) {
  v16h a;
  #pragma unroll
  for (int j = 0; j < 8; ++j) {
    int k0 = kbase + j;
    int k1 = kbase + 8 + j;
    a[j]     = (_Float16)((k0 < K) ? rowp[k0] : 0.f);
    a[8 + j] = (_Float16)((k1 < K) ? rowp[k1] : 0.f);
  }
  return a;
}

__device__ inline v16h load_b16(const _Float16* __restrict__ lds, int n, int stride, int kb2) {
  v16h zero = {};
  int nc = (n < SUB) ? n : 0;
  v16h b = *(const v16h*)(lds + nc * stride + kb2);  // 32B aligned (kb2 % 16 == 0)
  return (n < SUB) ? b : zero;
}

__global__ __launch_bounds__(256)
void glm_gemm_syn(const float* __restrict__ S_e, const float* __restrict__ S_i,
                  const float* __restrict__ C_syn_e, const float* __restrict__ C_syn_i,
                  float* __restrict__ synE, float* __restrict__ synI)
{
  __shared__ __align__(32) _Float16 ldsE[SUB][KE_PAD];  // 40 KB, zero-padded K
  __shared__ __align__(32) _Float16 ldsI[SUB][KI_PAD];  // 10 KB, zero-padded K

  const int tid = threadIdx.x;
  for (int idx = tid; idx < SUB * KE_PAD; idx += 256) {
    int s = idx / KE_PAD, k = idx % KE_PAD;
    ldsE[s][k] = (k < E_N) ? (_Float16)C_syn_e[s * E_N + k] : (_Float16)0.f;
  }
  for (int idx = tid; idx < SUB * KI_PAD; idx += 256) {
    int s = idx / KI_PAD, k = idx % KI_PAD;
    ldsI[s][k] = (k < I_N) ? (_Float16)C_syn_i[s * I_N + k] : (_Float16)0.f;
  }
  __syncthreads();

  const int lane  = tid & 31;
  const int wave  = tid >> 5;
  const int tileM = (blockIdx.x * 8 + wave) * 16;
  if (tileM >= T_TOT) return;

  const int m    = lane & 15;
  const int half = lane >> 4;          // K-half select (ISA A/B layouts)
  const int row  = tileM + m;
  const int nlo  = lane & 15;          // C/D column within 16-wide tile

  v8f accE0 = {}, accE1 = {}, accI0 = {}, accI1 = {};

  // ---- excitatory: K = 1000 (31 full chunks + guarded tail at 992) ----
  {
    const float* rowp = S_e + (size_t)row * E_N;
    for (int k0 = 0; k0 < 992; k0 += 32) {
      v16h a  = load_a_full(rowp, k0 + 16 * half);
      v16h b0 = load_b16(&ldsE[0][0], nlo,      KE_PAD, k0 + 16 * half);
      v16h b1 = load_b16(&ldsE[0][0], nlo + 16, KE_PAD, k0 + 16 * half);
      accE0 = __builtin_amdgcn_wmma_f32_16x16x32_f16(false, a, false, b0, (short)0, accE0, false, false);
      accE1 = __builtin_amdgcn_wmma_f32_16x16x32_f16(false, a, false, b1, (short)0, accE1, false, false);
    }
    v16h a  = load_a_guard(rowp, 992 + 16 * half, E_N);
    v16h b0 = load_b16(&ldsE[0][0], nlo,      KE_PAD, 992 + 16 * half);
    v16h b1 = load_b16(&ldsE[0][0], nlo + 16, KE_PAD, 992 + 16 * half);
    accE0 = __builtin_amdgcn_wmma_f32_16x16x32_f16(false, a, false, b0, (short)0, accE0, false, false);
    accE1 = __builtin_amdgcn_wmma_f32_16x16x32_f16(false, a, false, b1, (short)0, accE1, false, false);
  }

  // ---- inhibitory: K = 250 (8 guarded chunks; rows only 8B aligned -> scalar loads) ----
  {
    const float* rowp = S_i + (size_t)row * I_N;
    for (int k0 = 0; k0 < 256; k0 += 32) {
      v16h a  = load_a_guard(rowp, k0 + 16 * half, I_N);
      v16h b0 = load_b16(&ldsI[0][0], nlo,      KI_PAD, k0 + 16 * half);
      v16h b1 = load_b16(&ldsI[0][0], nlo + 16, KI_PAD, k0 + 16 * half);
      accI0 = __builtin_amdgcn_wmma_f32_16x16x32_f16(false, a, false, b0, (short)0, accI0, false, false);
      accI1 = __builtin_amdgcn_wmma_f32_16x16x32_f16(false, a, false, b1, (short)0, accI1, false, false);
    }
  }

  // C/D layout: VGPR j -> (M = j + 8*half, N = nlo [+16 for tile1])
  #pragma unroll
  for (int j = 0; j < 8; ++j) {
    int mm = j + 8 * half;
    size_t r = (size_t)(tileM + mm) * SUB;
    synE[r + nlo] = accE0[j];
    synI[r + nlo] = accI0[j];
    if (nlo < SUB - 16) {              // columns 16..19 only
      synE[r + 16 + nlo] = accE1[j];
      synI[r + 16 + nlo] = accI1[j];
    }
  }
}

// ---------------------------------------------------------------------------
// Kernel 2: build the four [SUB x 200] temporal kernels on device
// ---------------------------------------------------------------------------
__global__ void glm_build_kernels(const float* __restrict__ Tau_syn,
                                  const float* __restrict__ Delta_syn,
                                  const float* __restrict__ W_syn,
                                  const float* __restrict__ Tau_spk,
                                  const float* __restrict__ W_spk,
                                  const float* __restrict__ W_hist,
                                  float* __restrict__ ek, float* __restrict__ ik,
                                  float* __restrict__ sk, float* __restrict__ hk)
{
  int idx = blockIdx.x * blockDim.x + threadIdx.x;
  if (idx >= SUB * TN) return;
  int s = idx / TN, v = idx % TN;
  float t = (float)v;

  float te  = fmaxf(t - expf(Delta_syn[s * 2 + 0]), 0.f);
  float tte = te / expf(Tau_syn[s * 2 + 0]);
  ek[idx] = tte * expf(-tte) * expf(W_syn[s * 2 + 0]);

  float ti  = fmaxf(t - expf(Delta_syn[s * 2 + 1]), 0.f);
  float tti = ti / expf(Tau_syn[s * 2 + 1]);
  ik[idx] = -tti * expf(-tti) * expf(W_syn[s * 2 + 1]);

  float tts = t / expf(Tau_spk[s]);
  sk[idx] = tts * expf(-tts) * expf(W_spk[s]);

  // hist_kern = -(W_hist^2) @ COS_BASIS, basis recomputed from closed form
  const float PI_F = 3.14159265358979323846f;
  float raw = 3.0f * logf(t + 1.0f);
  float acc = 0.f;
  #pragma unroll
  for (int i = 0; i < NB; ++i) {
    float phi = 1.57079632679489662f * (float)i;
    float b = ((raw < phi - PI_F) || (raw > phi + PI_F)) ? 0.f
                                                         : (0.5f * cosf(raw - phi) + 0.5f);
    float w = W_hist[s * NB + i];
    acc = fmaf(w * w, b, acc);
  }
  hk[idx] = -acc;
}

// ---------------------------------------------------------------------------
// Kernel 3: LDS-tiled grouped causal convs + C_den mix + sigmoid
// out[t,s] = sigmoid( conv(synE,ek) + conv(synI,ik) + conv(Z,hk) + Theta[s]
//                     + sum_sp C_den[s][sp] * conv(Z,sk)[t,sp] )
// where conv(x,K)[t,s] = sum_{v=0..199} K[s][v] * x[t-1-v, s]  (zero for t-1-v<0)
// ---------------------------------------------------------------------------
#define ROWS (TILE + HIST)   // 456
#define RP   (SUB + 1)       // row pad 21 -> conflict-free LDS (gcd(21,64)=1)

__global__ __launch_bounds__(256)
void glm_conv_out(const float* __restrict__ synE, const float* __restrict__ synI,
                  const float* __restrict__ Z, const float* __restrict__ C_den,
                  const float* __restrict__ Theta,
                  const float* __restrict__ ek, const float* __restrict__ ik,
                  const float* __restrict__ sk, const float* __restrict__ hk,
                  float* __restrict__ outP)
{
  __shared__ float se[ROWS][RP];
  __shared__ float si[ROWS][RP];
  __shared__ float zl[ROWS][RP];
  __shared__ float zc[TILE][RP];
  __shared__ float cden[SUB][SUB];
  __shared__ float th[SUB];

  const int tid   = threadIdx.x;
  const int tile0 = blockIdx.x * TILE;

  // Stage tile + history (coalesced; zero-pad out-of-range rows)
  for (int idx = tid; idx < ROWS * SUB; idx += 256) {
    int r = idx / SUB, s = idx % SUB;
    int g = tile0 - HIST + r;
    bool ok = (g >= 0) && (g < T_TOT);
    size_t go = (size_t)(ok ? g : 0) * SUB + s;
    se[r][s] = ok ? synE[go] : 0.f;
    si[r][s] = ok ? synI[go] : 0.f;
    zl[r][s] = ok ? Z[go]    : 0.f;
  }
  for (int idx = tid; idx < SUB * SUB; idx += 256) cden[idx / SUB][idx % SUB] = C_den[idx];
  if (tid < SUB) th[tid] = Theta[tid];
  __syncthreads();

  // Pass 1: spike-coupling conv -> zc (needed cross-channel in pass 2)
  #pragma unroll 1
  for (int s = 0; s < SUB; ++s) {
    const float* skr = sk + s * TN;
    float aZ = 0.f;
    for (int v = 0; v < TN; ++v)
      aZ = fmaf(skr[v], zl[tid + (HIST - 1) - v][s], aZ);
    zc[tid][s] = aZ;
  }
  __syncthreads();

  // Pass 2: e/i/hist convs + dendritic mix + sigmoid
  const int tg = tile0 + tid;
  if (tg < T_TOT) {
    #pragma unroll 1
    for (int s = 0; s < SUB; ++s) {
      const float* ekr = ek + s * TN;
      const float* ikr = ik + s * TN;
      const float* hkr = hk + s * TN;
      float acc = th[s];
      for (int v = 0; v < TN; ++v) {
        int xi = tid + (HIST - 1) - v;
        acc = fmaf(ekr[v], se[xi][s], acc);
        acc = fmaf(ikr[v], si[xi][s], acc);
        acc = fmaf(hkr[v], zl[xi][s], acc);
      }
      float spk = 0.f;
      #pragma unroll
      for (int sp = 0; sp < SUB; ++sp)
        spk = fmaf(cden[s][sp], zc[tid][sp], spk);
      float x = acc + spk;
      outP[(size_t)tg * SUB + s] = 1.f / (1.f + expf(-x));
    }
  }
}

// ---------------------------------------------------------------------------
// Launcher. ws layout (floats): synE[2e6] | synI[2e6] | ek|ik|sk|hk [4000 each]
// => ~16.1 MB of scratch used.
// ---------------------------------------------------------------------------
extern "C" void kernel_launch(void* const* d_in, const int* in_sizes, int n_in,
                              void* d_out, int out_size, void* d_ws, size_t ws_size,
                              hipStream_t stream) {
  const float* S_e       = (const float*)d_in[0];
  const float* S_i       = (const float*)d_in[1];
  const float* Z         = (const float*)d_in[2];
  const float* C_den     = (const float*)d_in[3];
  const float* C_syn_e   = (const float*)d_in[4];
  const float* C_syn_i   = (const float*)d_in[5];
  const float* Tau_syn   = (const float*)d_in[6];
  const float* Delta_syn = (const float*)d_in[7];
  const float* W_syn     = (const float*)d_in[8];
  const float* Tau_spk   = (const float*)d_in[9];
  const float* W_spk     = (const float*)d_in[10];
  const float* W_hist    = (const float*)d_in[11];
  const float* Theta     = (const float*)d_in[12];
  float* outP = (float*)d_out;

  float* ws   = (float*)d_ws;
  float* synE = ws;
  float* synI = ws + (size_t)T_TOT * SUB;
  float* ek   = ws + (size_t)2 * T_TOT * SUB;
  float* ik   = ek + SUB * TN;
  float* sk   = ik + SUB * TN;
  float* hk   = sk + SUB * TN;

  const int rowTiles  = T_TOT / 16;                  // 6250
  const int gemmBlks  = (rowTiles + 7) / 8;          // 782
  glm_gemm_syn<<<gemmBlks, 256, 0, stream>>>(S_e, S_i, C_syn_e, C_syn_i, synE, synI);

  glm_build_kernels<<<(SUB * TN + 255) / 256, 256, 0, stream>>>(
      Tau_syn, Delta_syn, W_syn, Tau_spk, W_spk, W_hist, ek, ik, sk, hk);

  const int convBlks = (T_TOT + TILE - 1) / TILE;    // 391
  glm_conv_out<<<convBlks, 256, 0, stream>>>(
      synE, synI, Z, C_den, Theta, ek, ik, sk, hk, outP);
}
